// TSPBlock_38714835206279
// MI455X (gfx1250) — matrix operand order repaired and checked by reading, not compiled, not run on hardware
//
#include <hip/hip_runtime.h>
#include <hip/hip_bf16.h>
#include <math.h>

// ---------------------------------------------------------------------------
// 3-layer GAT for MI455X (gfx1250, wave32, WMMA).
//   GEMMs:  v_wmma_f32_16x16x32_bf16, one 16x16 tile per wave, 8 waves/block.
//   Edge softmax: global_atomic_max_u32 on order-preserving float encoding,
//                 then global_atomic_add_f32 scatter for den / weighted agg.
//   BN: block-local channel partials + one atomicAdd per channel per block.
// All tensors (<=26MB each) fit in the 192MB L2, so the pipeline is
// L2-resident; bf16 WMMA is the highest-throughput matrix path on CDNA5.
// ---------------------------------------------------------------------------

#define GN 50000      // nodes
#define GE 800000     // edges
#define HID 128       // H*D
#define NH 8          // heads
#define HD 16         // per-head dim

typedef __attribute__((ext_vector_type(16))) __bf16 v16bf;
typedef __attribute__((ext_vector_type(8)))  __bf16 v8bf;
typedef __attribute__((ext_vector_type(8)))  float  v8f;

static inline unsigned cdivu(size_t a, size_t b) { return (unsigned)((a + b - 1) / b); }

// ---------------- small helpers ----------------

__device__ __forceinline__ unsigned short f32_to_bf16_rne(float f) {
  unsigned u = __float_as_uint(f);
  u += 0x7FFFu + ((u >> 16) & 1u);       // round-to-nearest-even
  return (unsigned short)(u >> 16);
}

// order-preserving encoding: unsigned compare order == float compare order
__device__ __forceinline__ unsigned enc_f32(float f) {
  unsigned u = __float_as_uint(f);
  return (u & 0x80000000u) ? ~u : (u | 0x80000000u);
}
__device__ __forceinline__ float dec_f32(unsigned u) {
  unsigned v = (u & 0x80000000u) ? (u & 0x7FFFFFFFu) : ~u;
  return __uint_as_float(v);
}

__global__ void fill_u32_kernel(unsigned* __restrict__ p, unsigned v, size_t n) {
  size_t i = (size_t)blockIdx.x * blockDim.x + threadIdx.x;
  if (i < n) p[i] = v;
}

// out[i] = bf16(a[i] (+ b[i]))   (b may be null: used for the h1+h2 skip)
__global__ void to_bf16_kernel(const float* __restrict__ a, const float* __restrict__ b,
                               unsigned short* __restrict__ o, size_t n) {
  size_t i = (size_t)blockIdx.x * blockDim.x + threadIdx.x;
  if (i >= n) return;
  float v = a[i] + (b ? b[i] : 0.0f);
  o[i] = f32_to_bf16_rne(v);
}

// wt[n*128+k] = bf16(W[k*128+n])  (transpose 128x128 weight for coalesced B loads)
__global__ void wt_bf16_kernel(const float* __restrict__ W, unsigned short* __restrict__ wt) {
  int i = blockIdx.x * blockDim.x + threadIdx.x;
  if (i >= HID * HID) return;
  int n = i >> 7, k = i & 127;
  wt[i] = f32_to_bf16_rne(W[k * HID + n]);
}

// ---------------- GEMM: h[N,128] = xb[N,128] @ W (via bf16 WMMA) ----------------
// One 16x16 C tile per wave; wave w of the block owns column tile w (8 waves
// cover all 128 cols); blockIdx.x is the row tile (50000/16 = 3125 exactly).
// A consistent K-permutation is applied to both A and B fragments so that
// each lane's A fragment is one contiguous 32B load and its B fragment is
// one contiguous 32B load from the transposed weight.
__global__ __launch_bounds__(256) void gemm_wmma_kernel(
    const unsigned short* __restrict__ xb,   // [N,128] bf16 row-major
    const unsigned short* __restrict__ wt,   // [128,128] bf16, wt[n][k] = W[k][n]
    float* __restrict__ h)                   // [N,128] f32
{
  const int wave = threadIdx.x >> 5;
  const int lane = threadIdx.x & 31;
  const int hi   = lane >> 4;        // half-wave select
  const int l15  = lane & 15;
  const int rt   = blockIdx.x;       // row tile
  const int arow = rt * 16 + l15;    // A row held by this lane
  const int bcol = wave * 16 + l15;  // B/C column held by this lane

  const __bf16* ap = reinterpret_cast<const __bf16*>(xb) + (size_t)arow * HID;
  const __bf16* bp = reinterpret_cast<const __bf16*>(wt) + (size_t)bcol * HID;

  v8f c = {};
  #pragma unroll
  for (int kb = 0; kb < HID; kb += 32) {
    // A fragment: logical K = kb + hi*16 + (0..15)  -> one 32B contiguous load
    v16bf a = *reinterpret_cast<const v16bf*>(ap + kb + hi * 16);
    // B fragment: logical K = kb + hi*16 + (0..15)  -> one 32B contiguous load
    v16bf b = *reinterpret_cast<const v16bf*>(bp + kb + hi * 16);
    c = __builtin_amdgcn_wmma_f32_16x16x32_bf16(
        /*neg_a=*/false, a, /*neg_b=*/false, b,
        /*c_mod=*/(short)0, c, /*reuse_a=*/false, /*reuse_b=*/false);
  }

  // C/D layout: VGPR i holds row M = i + hi*8, col = lane&15 (per 16x16 f32 tile)
  float* hout = h + (size_t)(rt * 16 + hi * 8) * HID + bcol;
  #pragma unroll
  for (int i = 0; i < 8; ++i) hout[(size_t)i * HID] = c[i];
}

// ---------------- attention scores: es/ed[n,h] = <h[n,h,:], a[h,:D]/a[h,D:]> ----
__global__ void scores_kernel(const float* __restrict__ h, const float* __restrict__ a,
                              float* __restrict__ es, float* __restrict__ ed) {
  int idx = blockIdx.x * blockDim.x + threadIdx.x;   // node*8 + head
  if (idx >= GN * NH) return;
  int node = idx >> 3, hd = idx & 7;
  const float* hp = h + (size_t)node * HID + hd * HD;
  const float* as = a + hd * 2 * HD;
  float s = 0.f, d = 0.f;
  #pragma unroll
  for (int i = 0; i < HD; ++i) { s += hp[i] * as[i]; d += hp[i] * as[HD + i]; }
  es[idx] = s; ed[idx] = d;
}

// ---------------- edge pass 1: per-dst segment max (stable softmax) ------------
__global__ void edge_max_kernel(const int* __restrict__ src, const int* __restrict__ dst,
                                const float* __restrict__ es, const float* __restrict__ ed,
                                unsigned* __restrict__ menc) {
  int idx = blockIdx.x * blockDim.x + threadIdx.x;   // edge*8 + head
  if (idx >= GE * NH) return;
  int e = idx >> 3, hd = idx & 7;
  int s = src[e], d = dst[e];
  float v = es[s * NH + hd] + ed[d * NH + hd];
  v = (v >= 0.f) ? v : 0.2f * v;                     // LeakyReLU(0.2)
  atomicMax(menc + d * NH + hd, enc_f32(v));
}

// decode in place: menc (uint encoding) -> float max; empty segment (-inf) -> 0
__global__ void decode_max_kernel(unsigned* __restrict__ m, int n) {
  int i = blockIdx.x * blockDim.x + threadIdx.x;
  if (i >= n) return;
  float v = dec_f32(m[i]);
  unsigned bits = __float_as_uint(v);
  bool finite = ((bits >> 23) & 0xFFu) != 0xFFu;
  m[i] = __float_as_uint(finite ? v : 0.0f);
}

// ---------------- edge pass 2: w = exp(e-m), den += w, agg += h[src]*w ---------
__global__ void edge_acc_kernel(const int* __restrict__ src, const int* __restrict__ dst,
                                const float* __restrict__ es, const float* __restrict__ ed,
                                const float* __restrict__ m, const float* __restrict__ h,
                                float* __restrict__ den, float* __restrict__ agg) {
  int idx = blockIdx.x * blockDim.x + threadIdx.x;   // edge*8 + head
  if (idx >= GE * NH) return;
  int e = idx >> 3, hd = idx & 7;
  int s = src[e], d = dst[e];
  float v = es[s * NH + hd] + ed[d * NH + hd];
  v = (v >= 0.f) ? v : 0.2f * v;
  float w = expf(v - m[d * NH + hd]);
  atomicAdd(den + d * NH + hd, w);
  const float* hs = h + (size_t)s * HID + hd * HD;
  float* ag = agg + (size_t)d * HID + hd * HD;
  #pragma unroll
  for (int i = 0; i < HD; ++i) atomicAdd(ag + i, hs[i] * w);
}

// ---------------- normalize by den + channel stats for BatchNorm ---------------
// 128 threads = 128 channels; each block handles 32 nodes, one atomic per
// channel per block into the global sums / sums-of-squares.
__global__ __launch_bounds__(128) void norm_stats_kernel(
    float* __restrict__ agg, const float* __restrict__ den,
    float* __restrict__ sums, float* __restrict__ sumsq) {
  int c = threadIdx.x;
  int base = blockIdx.x * 32;
  float s = 0.f, s2 = 0.f;
  for (int j = 0; j < 32; ++j) {
    int node = base + j;
    if (node >= GN) break;
    size_t off = (size_t)node * HID + c;
    float y = agg[off] / (den[node * NH + (c >> 4)] + 1e-16f);
    agg[off] = y;                       // y now holds the GAT layer output
    s += y; s2 += y * y;
  }
  atomicAdd(&sums[c], s);
  atomicAdd(&sumsq[c], s2);
}

// ---------------- apply BatchNorm (+ optional ELU) -----------------------------
__global__ void bn_apply_kernel(const float* __restrict__ y,
                                const float* __restrict__ sums, const float* __restrict__ sumsq,
                                const float* __restrict__ g, const float* __restrict__ b,
                                float* __restrict__ out, int do_elu) {
  int idx = blockIdx.x * blockDim.x + threadIdx.x;
  if (idx >= GN * HID) return;
  int c = idx & 127;
  const float inv_n = 1.0f / (float)GN;
  float mean = sums[c] * inv_n;
  float var  = sumsq[c] * inv_n - mean * mean;      // biased, matches jnp.var
  float v = g[c] * (y[idx] - mean) * rsqrtf(var + 1e-5f) + b[c];
  if (do_elu) v = (v > 0.f) ? v : (expf(v) - 1.0f); // jax.nn.elu
  out[idx] = v;
}

// ---------------- one full GAT layer -------------------------------------------
static void run_layer(const float* xin, const float* xskip,
                      const float* W, const float* a,
                      const float* g, const float* b,
                      float* out, int do_elu,
                      unsigned short* xb, unsigned short* wt, float* h,
                      float* es, float* ed, unsigned* menc, float* den, float* agg,
                      float* sums,  // 256 floats: [0:128) sums, [128:256) sumsq
                      const int* src, const int* dst, hipStream_t stream) {
  const size_t NF = (size_t)GN * HID;
  to_bf16_kernel<<<cdivu(NF, 256), 256, 0, stream>>>(xin, xskip, xb, NF);
  wt_bf16_kernel<<<cdivu(HID * HID, 256), 256, 0, stream>>>(W, wt);
  gemm_wmma_kernel<<<GN / 16, 256, 0, stream>>>(xb, wt, h);
  scores_kernel<<<cdivu((size_t)GN * NH, 256), 256, 0, stream>>>(h, a, es, ed);

  fill_u32_kernel<<<cdivu((size_t)GN * NH, 256), 256, 0, stream>>>(menc, 0x007FFFFFu, (size_t)GN * NH); // enc(-inf)
  fill_u32_kernel<<<cdivu((size_t)GN * NH, 256), 256, 0, stream>>>((unsigned*)den, 0u, (size_t)GN * NH);
  fill_u32_kernel<<<cdivu(NF, 256), 256, 0, stream>>>((unsigned*)agg, 0u, NF);
  fill_u32_kernel<<<1, 256, 0, stream>>>((unsigned*)sums, 0u, 256);

  edge_max_kernel<<<cdivu((size_t)GE * NH, 256), 256, 0, stream>>>(src, dst, es, ed, menc);
  decode_max_kernel<<<cdivu((size_t)GN * NH, 256), 256, 0, stream>>>(menc, GN * NH);
  edge_acc_kernel<<<cdivu((size_t)GE * NH, 256), 256, 0, stream>>>(
      src, dst, es, ed, (const float*)menc, h, den, agg);

  norm_stats_kernel<<<cdivu(GN, 32), 128, 0, stream>>>(agg, den, sums, sums + 128);
  bn_apply_kernel<<<cdivu(NF, 256), 256, 0, stream>>>(agg, sums, sums + 128, g, b, out, do_elu);
}

// ---------------- entry point ---------------------------------------------------
extern "C" void kernel_launch(void* const* d_in, const int* in_sizes, int n_in,
                              void* d_out, int out_size, void* d_ws, size_t ws_size,
                              hipStream_t stream) {
  const float* x  = (const float*)d_in[0];
  const int*   ei = (const int*)d_in[1];          // [2,E] int32
  const float* W1 = (const float*)d_in[2];  const float* a1 = (const float*)d_in[3];
  const float* W2 = (const float*)d_in[4];  const float* a2 = (const float*)d_in[5];
  const float* W3 = (const float*)d_in[6];  const float* a3 = (const float*)d_in[7];
  const float* g1 = (const float*)d_in[8];  const float* b1 = (const float*)d_in[9];
  const float* g2 = (const float*)d_in[10]; const float* b2 = (const float*)d_in[11];
  const float* g3 = (const float*)d_in[12]; const float* b3 = (const float*)d_in[13];
  const int* src = ei;
  const int* dst = ei + GE;

  // carve 256B-aligned workspace regions (~120 MB total)
  char* p = (char*)d_ws;
  auto carve = [&](size_t bytes) -> char* {
    char* r = p;
    p += (bytes + 255) & ~(size_t)255;
    return r;
  };
  unsigned short* xb  = (unsigned short*)carve((size_t)GN * HID * 2); // bf16 GEMM input
  unsigned short* wt  = (unsigned short*)carve((size_t)HID * HID * 2);// bf16 W^T
  float* h    = (float*)carve((size_t)GN * HID * 4);                  // GEMM output
  float* es   = (float*)carve((size_t)GN * NH * 4);
  float* ed   = (float*)carve((size_t)GN * NH * 4);
  unsigned* menc = (unsigned*)carve((size_t)GN * NH * 4);             // max enc -> float
  float* den  = (float*)carve((size_t)GN * NH * 4);
  float* agg  = (float*)carve((size_t)GN * HID * 4);
  float* out1 = (float*)carve((size_t)GN * HID * 4);                  // ELU(BN(layer1))
  float* out2 = (float*)carve((size_t)GN * HID * 4);                  // ELU(BN(layer2))
  float* sums = (float*)carve(256 * 4);                               // sums | sumsq

  // layer 1: x -> out1 (BN + ELU)
  run_layer(x, nullptr, W1, a1, g1, b1, out1, 1,
            xb, wt, h, es, ed, menc, den, agg, sums, src, dst, stream);
  // layer 2: out1 -> out2 (BN + ELU)
  run_layer(out1, nullptr, W2, a2, g2, b2, out2, 1,
            xb, wt, h, es, ed, menc, den, agg, sums, src, dst, stream);
  // layer 3: (out1 + out2) -> d_out (BN only); skip-add fused into bf16 convert
  run_layer(out1, out2, W3, a3, g3, b3, (float*)d_out, 0,
            xb, wt, h, es, ed, menc, den, agg, sums, src, dst, stream);
}
